// Scorer_33380485825322
// MI455X (gfx1250) — matrix operand order repaired and checked by reading, not compiled, not run on hardware
//
#include <hip/hip_runtime.h>

// Problem constants (match reference)
#define BATCH 32
#define TLEN  2048
#define DIN   1024
#define DHID  2048
#define NTAG  2

#define BLOCK_ROWS 128                 // 8 waves x 16 private rows
#define NCHUNK     (TLEN / 16)         // 128 per-sample 16-row chunks
#define LDS_PITCH  1032                // 1024 + 8 bf16 pad -> conflict-free b128

typedef __attribute__((ext_vector_type(16))) __bf16 v16bf;
typedef __attribute__((ext_vector_type(8)))  float  v8f;
typedef __attribute__((ext_vector_type(4)))  unsigned short usv4;

union BfFrag {
    v16bf v;
    uint4 q[2];
};

// fp32 -> bf16 round-to-nearest-even
__device__ __forceinline__ unsigned short f2bf(float f) {
    unsigned int u = __float_as_uint(f);
    unsigned int r = u + 0x7FFFu + ((u >> 16) & 1u);
    return (unsigned short)(r >> 16);
}

// ---------------------------------------------------------------------------
// Kernel 1: convert W1 (fp32 [DHID,DIN]) -> bf16 in ws (L2-resident operand).
// ---------------------------------------------------------------------------
__global__ void prep_kernel(const float* __restrict__ W1,
                            unsigned short* __restrict__ w1bf) {
    int g = blockIdx.x * blockDim.x + threadIdx.x;
    if (g < DHID * DIN) w1bf[g] = f2bf(W1[g]);
}

// ---------------------------------------------------------------------------
// Kernel 2: fused  relu(h @ W1^T + b1)  +  per-row-masked column sums.
// grid = (TLEN/128, BATCH), block = 256 = 8 waves.
// Block stages a 128x1024 bf16 h-tile in LDS (264 KB of the WGP's 320 KB).
// Wave w owns rows w*16..w*16+15 and sweeps ALL 2048 columns, so W1-bf16 is
// streamed from L2 once per 128 rows (~2 GB total) while h comes from HBM
// exactly once (256 MB). Partial sums go to ws with plain stores
// (fixed-order reduction later => fully deterministic, no atomics).
// ---------------------------------------------------------------------------
__global__ void __launch_bounds__(256)
gemm_pool_kernel(const float* __restrict__ h,
                 const int* __restrict__ text_len,
                 const float* __restrict__ b1,
                 const unsigned short* __restrict__ w1bf,
                 float* __restrict__ part) {
    const int b      = blockIdx.y;
    const int t_base = blockIdx.x * BLOCK_ROWS;
    const int len    = text_len[b];

    __shared__ unsigned short As[BLOCK_ROWS][LDS_PITCH];   // 264,192 bytes

    const int tid  = threadIdx.x;
    const int lane = tid & 31;
    const int wave = tid >> 5;

    // ---- stage + convert h[b, t_base .. t_base+127, :] into LDS ----
    // (rows past len are loaded too; they are masked after relu, which is
    //  required anyway because relu(b1) != 0.)
    const float* hbase = h + ((size_t)b * TLEN + t_base) * (size_t)DIN;
    for (int i = tid; i < BLOCK_ROWS * (DIN / 4); i += 256) {  // 32768 float4s
        int row = i >> 8;            // 256 float4 per row
        int c4  = i & 255;
        float4 v = ((const float4*)(hbase + (size_t)row * DIN))[c4];
        usv4 s;
        s[0] = f2bf(v.x); s[1] = f2bf(v.y); s[2] = f2bf(v.z); s[3] = f2bf(v.w);
        *(usv4*)&As[row][c4 * 4] = s;
    }
    __syncthreads();

    const int r0       = wave * 16;                 // this wave's row sub-tile
    const int chunk    = blockIdx.x * 8 + wave;     // global 16-row chunk id
    const int m_row    = lane & 15;                 // A row / B column selector
    const int a_koff   = (lane >> 4) * 8;           // A half-wave K split
    const int b_koff   = (lane >> 4) * 16;          // unused; B loads also use a_koff
    const int m_base   = (lane >> 4) * 8;           // C rows 0..7 / 8..15
    const int wave_t0  = t_base + r0;
    const bool valid   = (wave_t0 < len);

    float* pslot = part + ((size_t)b * NCHUNK + chunk) * DHID;

    if (valid) {
        const unsigned short* arow = &As[r0 + m_row][0];
        for (int n_tile = 0; n_tile < DHID; n_tile += 16) {
            const int n = n_tile + m_row;
            const unsigned short* wrow = w1bf + (size_t)n * DIN;

            // issue bias load BEFORE the K-loop so its latency hides behind
            // the 32 WMMAs instead of stalling the epilogue every n-tile
            const float bias = b1[n];

            v8f c0 = {0.f,0.f,0.f,0.f,0.f,0.f,0.f,0.f};
            v8f c1 = {0.f,0.f,0.f,0.f,0.f,0.f,0.f,0.f};

            for (int k = 0; k < DIN; k += 64) {
                BfFrag a, bb;
                // even k-step -> c0
                a.q[0]  = *(const uint4*)(arow + k + a_koff);
                a.q[1]  = *(const uint4*)(arow + k + 16 + a_koff);
                bb.q[0] = *(const uint4*)(wrow + k + a_koff);
                bb.q[1] = *(const uint4*)(wrow + k + 16 + a_koff);
                c0 = __builtin_amdgcn_wmma_f32_16x16x32_bf16(
                        false, a.v, false, bb.v, (short)0, c0, false, false);
                // odd k-step -> c1 (independent chain)
                a.q[0]  = *(const uint4*)(arow + k + 32 + a_koff);
                a.q[1]  = *(const uint4*)(arow + k + 48 + a_koff);
                bb.q[0] = *(const uint4*)(wrow + k + 32 + a_koff);
                bb.q[1] = *(const uint4*)(wrow + k + 48 + a_koff);
                c1 = __builtin_amdgcn_wmma_f32_16x16x32_bf16(
                        false, a.v, false, bb.v, (short)0, c1, false, false);
            }
            v8f c = c0 + c1;

            // bias + relu + per-row mask, fold the 16 rows of this column
            float partial = 0.0f;
#pragma unroll
            for (int r = 0; r < 8; ++r) {
                int t = wave_t0 + m_base + r;
                float v = fmaxf(c[r] + bias, 0.0f);
                partial += (t < len) ? v : 0.0f;
            }
            partial += __shfl_xor(partial, 16);   // fold rows 0-7 with 8-15
            if (lane < 16) pslot[n] = partial;    // plain store, deterministic
        }
    } else {
        // fully-masked chunk: ws is poisoned, so write zeros
        for (int n_tile = 0; n_tile < DHID; n_tile += 16)
            if (lane < 16) pslot[n_tile + lane] = 0.0f;
    }
}

// ---------------------------------------------------------------------------
// Kernel 3: pooled[b,n] = sum_c part[b,c,n] (fixed order); then
//           score[b,tag] = dot(pooled[b,:], W2[tag,:]) / len[b] + b2[tag]
// ---------------------------------------------------------------------------
__global__ void __launch_bounds__(256)
score_kernel(const float* __restrict__ part,
             const float* __restrict__ W2,
             const float* __restrict__ b2,
             const int* __restrict__ text_len,
             float* __restrict__ out) {
    const int b   = blockIdx.x;
    const int tid = threadIdx.x;
    __shared__ float red0[256];
    __shared__ float red1[256];

    float acc0 = 0.0f, acc1 = 0.0f;
    const float* pb = part + (size_t)b * NCHUNK * DHID;
    for (int n = tid; n < DHID; n += 256) {
        float pooled = 0.0f;
        for (int c = 0; c < NCHUNK; ++c)          // fixed ascending order
            pooled += pb[(size_t)c * DHID + n];
        acc0 += pooled * W2[n];
        acc1 += pooled * W2[DHID + n];
    }
    red0[tid] = acc0; red1[tid] = acc1;
    __syncthreads();
#pragma unroll
    for (int s = 128; s > 0; s >>= 1) {
        if (tid < s) { red0[tid] += red0[tid + s]; red1[tid] += red1[tid + s]; }
        __syncthreads();
    }
    if (tid == 0) {
        float inv = 1.0f / (float)text_len[b];
        out[b * NTAG + 0] = red0[0] * inv + b2[0];
        out[b * NTAG + 1] = red1[0] * inv + b2[1];
    }
}

// ---------------------------------------------------------------------------
// Launch: inputs = { h, text_len, W1, b1, W2, b2 }, output fp32 [32,2]
// ws layout: [ W1 bf16: 4 MB ][ partials: B*128*2048 f32 = 32 MB ]
// ---------------------------------------------------------------------------
extern "C" void kernel_launch(void* const* d_in, const int* in_sizes, int n_in,
                              void* d_out, int out_size, void* d_ws, size_t ws_size,
                              hipStream_t stream) {
    const float* h        = (const float*)d_in[0];
    const int*   text_len = (const int*)d_in[1];
    const float* W1       = (const float*)d_in[2];
    const float* b1       = (const float*)d_in[3];
    const float* W2       = (const float*)d_in[4];
    const float* b2       = (const float*)d_in[5];
    float* out            = (float*)d_out;

    unsigned short* w1bf  = (unsigned short*)d_ws;
    float* part           = (float*)((char*)d_ws + (size_t)DHID * DIN * sizeof(unsigned short));

    {   // 1) convert W1 -> bf16
        int n = DHID * DIN;
        prep_kernel<<<(n + 255) / 256, 256, 0, stream>>>(W1, w1bf);
    }
    {   // 2) fused GEMM + relu + masked pooling (partials)
        dim3 grid(TLEN / BLOCK_ROWS, BATCH);
        gemm_pool_kernel<<<grid, 256, 0, stream>>>(h, text_len, b1, w1bf, part);
    }
    // 3) deterministic reduction + classifier head
    score_kernel<<<BATCH, 256, 0, stream>>>(part, W2, b2, text_len, out);
}
